// SelfAttentionModel_5695126635034
// MI455X (gfx1250) — compile-verified, matrix operand order checked by reading, not run on hardware
//
#include <hip/hip_runtime.h>
#include <hip/hip_bf16.h>

typedef _Float16 f16;
typedef __attribute__((ext_vector_type(16))) _Float16 v16h;
typedef __attribute__((ext_vector_type(8)))  float    v8f;

#define B_ 4
#define T_ 1024
#define V_ 50257
#define E_ 1024
#define H_ 16
#define D_ 64
#define L_ 8

#define GF_BIAS 1
#define GF_RELU 2
#define GF_RES  4

// ---------------------------------------------------------------------------
// Generic batched GEMM, C = A * B^T (+bias)(+residual)(relu), via WMMA f16.
// A   : [batch][M,K] f16 (row-major, lda = K), batch stride aB
// Bt  : [batch][N,K] f16 (i.e. B stored transposed),       batch stride bB
// C   : element (m,n) goes to  coff + m*ldc + n  where
//       coff = (z/czDiv)*czOuter + (z%czDiv)*czInner   (z = blockIdx.z)
// Each wave computes a 64x64 tile (4x4 WMMA subtiles -> 16 v_wmma per K=32
// step against 8 fragment loads); 2x2 waves/block -> 128x128 block tile.
// ---------------------------------------------------------------------------
__global__ __launch_bounds__(128)
void gemm_wmma_f16(const f16* __restrict__ A, const f16* __restrict__ Bt,
                   const float* __restrict__ bias, const float* __restrict__ residual,
                   float* __restrict__ Cf, f16* __restrict__ Ch,
                   int M, int N, int K, int ldc,
                   long long aB, long long bB,
                   int czDiv, long long czOuter, long long czInner,
                   int flags)
{
    const int z = blockIdx.z;
    const f16* Az = A + (long long)z * aB;
    const f16* Bz = Bt + (long long)z * bB;
    const long long coff = (long long)(z / czDiv) * czOuter
                         + (long long)(z % czDiv) * czInner;

    const int lane = threadIdx.x & 31;
    const int wave = threadIdx.x >> 5;
    const int wm = wave & 1;          // 2 waves along M
    const int wn = wave >> 1;         // 2 waves along N
    const int m0 = blockIdx.x * 128 + wm * 64;
    const int n0 = blockIdx.y * 128 + wn * 64;

    const int lr = lane & 15;
    const int ks = (lane >> 4) << 4;  // K-half select: 0 or 16

    // A fragment row pointers (four 16-row subtiles), clamped for safety
    const f16* ap[4];
#pragma unroll
    for (int i = 0; i < 4; ++i) {
        int r = m0 + i * 16 + lr;
        if (r > M - 1) r = M - 1;
        ap[i] = Az + (long long)r * K + ks;
    }

    // B fragment column pointers (four 16-col subtiles), clamped for ragged N
    const f16* bp[4];
#pragma unroll
    for (int j = 0; j < 4; ++j) {
        int nc = n0 + j * 16 + lr;
        if (nc > N - 1) nc = N - 1;
        bp[j] = Bz + (long long)nc * K + ks;
    }

    v8f acc[4][4] = {};

    for (int k0 = 0; k0 < K; k0 += 32) {
        v16h a[4], b[4];
#pragma unroll
        for (int i = 0; i < 4; ++i) a[i] = *(const v16h*)(ap[i] + k0);
#pragma unroll
        for (int j = 0; j < 4; ++j) b[j] = *(const v16h*)(bp[j] + k0);
#pragma unroll
        for (int i = 0; i < 4; ++i)
#pragma unroll
            for (int j = 0; j < 4; ++j)
                acc[i][j] = __builtin_amdgcn_wmma_f32_16x16x32_f16(
                    false, a[i], false, b[j], (short)0, acc[i][j], false, false);
    }

    // Epilogue. C layout: VGPR r -> M = r + 8*(lane>>4), N = lane&15.
#pragma unroll
    for (int i = 0; i < 4; ++i) {
        const int mb = m0 + i * 16 + (lane >> 4) * 8;
#pragma unroll
        for (int j = 0; j < 4; ++j) {
            const int nn = n0 + j * 16 + lr;
            if (nn >= N) continue;
            const float bv = (flags & GF_BIAS) ? bias[nn] : 0.0f;
#pragma unroll
            for (int r = 0; r < 8; ++r) {
                const int mm = mb + r;
                if (mm >= M) continue;
                float val = acc[i][j][r] + bv;
                if (flags & GF_RELU) val = val > 0.0f ? val : 0.0f;
                const long long off = coff + (long long)mm * ldc + nn;
                if (flags & GF_RES) val += residual[off];
                if (Cf) Cf[off] = val;
                if (Ch) Ch[off] = (f16)val;
            }
        }
    }
}

// ---------------------------------------------------------------------------
// x[b,t,:] = tok_emb[idx[b,t],:] * pos_emb[t,:]   (fp32)
// ---------------------------------------------------------------------------
__global__ __launch_bounds__(256)
void embed_mul(const int* __restrict__ idx, const float* __restrict__ tok,
               const float* __restrict__ pos, float* __restrict__ x)
{
    const long long gid = (long long)blockIdx.x * 256 + threadIdx.x;
    if (gid >= (long long)B_ * T_ * E_) return;
    const int e = (int)(gid % E_);
    const long long m = gid / E_;
    const int t = (int)(m % T_);
    const int tid_tok = idx[m];
    x[gid] = tok[(long long)tid_tok * E_ + e] * pos[(long long)t * E_ + e];
}

// ---------------------------------------------------------------------------
// LayerNorm over E=1024, fp32 in -> f16 out. One block (256 thr) per row.
// ---------------------------------------------------------------------------
__global__ __launch_bounds__(256)
void layernorm_f16(const float* __restrict__ x, const float* __restrict__ w,
                   const float* __restrict__ b, f16* __restrict__ y)
{
    const long long row = blockIdx.x;
    const float* xr = x + row * E_;
    f16* yr = y + row * E_;
    __shared__ float red[256];
    const int tid = threadIdx.x;

    float v[4]; float s = 0.0f;
#pragma unroll
    for (int i = 0; i < 4; ++i) { v[i] = xr[tid + i * 256]; s += v[i]; }
    red[tid] = s; __syncthreads();
    for (int st = 128; st; st >>= 1) { if (tid < st) red[tid] += red[tid + st]; __syncthreads(); }
    const float mu = red[0] * (1.0f / E_); __syncthreads();

    float q = 0.0f;
#pragma unroll
    for (int i = 0; i < 4; ++i) { const float d = v[i] - mu; q += d * d; }
    red[tid] = q; __syncthreads();
    for (int st = 128; st; st >>= 1) { if (tid < st) red[tid] += red[tid + st]; __syncthreads(); }
    const float rstd = rsqrtf(red[0] * (1.0f / E_) + 1e-5f);

#pragma unroll
    for (int i = 0; i < 4; ++i) {
        const int c = tid + i * 256;
        yr[c] = (f16)((v[i] - mu) * rstd * w[c] + b[c]);
    }
}

// ---------------------------------------------------------------------------
// Split [B*T, E] f16 into per-head layout.
//   transposeV==0 : dst[b,h,t,d]  (for q,k)
//   transposeV==1 : dst[b,h,d,t]  (for v, so att@v hits contiguous B-frags)
// ---------------------------------------------------------------------------
__global__ __launch_bounds__(256)
void split_heads(const f16* __restrict__ src, f16* __restrict__ dst, int transposeV)
{
    const long long gid = (long long)blockIdx.x * 256 + threadIdx.x;
    if (gid >= (long long)B_ * T_ * E_) return;
    const int e = (int)(gid % E_);
    const long long m = gid / E_;
    const int t = (int)(m % T_);
    const int b = (int)(m / T_);
    const int h = e / D_;
    const int d = e % D_;
    const long long o = transposeV
        ? (((long long)(b * H_ + h) * D_ + d) * T_ + t)
        : (((long long)(b * H_ + h) * T_ + t) * D_ + d);
    dst[o] = src[gid];
}

// ---------------------------------------------------------------------------
// In-place causal softmax over f16 score rows (scale applied here).
// One block per (b,h,tq) row; T=1024, 4 cols/thread.
// ---------------------------------------------------------------------------
__global__ __launch_bounds__(256)
void softmax_causal(f16* __restrict__ att, float scale)
{
    const long long row = blockIdx.x;          // over B*H*T rows
    const int tq = (int)(row % T_);
    f16* p = att + row * T_;
    __shared__ float red[256];
    const int tid = threadIdx.x;

    float v[4]; float mx = -1e30f;
#pragma unroll
    for (int i = 0; i < 4; ++i) {
        const int c = tid + i * 256;
        float s = (float)p[c] * scale;
        if (c > tq) s = -1e30f;                // causal mask
        v[i] = s; mx = fmaxf(mx, s);
    }
    red[tid] = mx; __syncthreads();
    for (int st = 128; st; st >>= 1) { if (tid < st) red[tid] = fmaxf(red[tid], red[tid + st]); __syncthreads(); }
    mx = red[0]; __syncthreads();

    float sum = 0.0f;
#pragma unroll
    for (int i = 0; i < 4; ++i) { v[i] = expf(v[i] - mx); sum += v[i]; }
    red[tid] = sum; __syncthreads();
    for (int st = 128; st; st >>= 1) { if (tid < st) red[tid] += red[tid + st]; __syncthreads(); }
    const float inv = 1.0f / red[0];

#pragma unroll
    for (int i = 0; i < 4; ++i) p[tid + i * 256] = (f16)(v[i] * inv);
}

// ---------------------------------------------------------------------------
// Tiled fp32 -> f16 transpose: dst[C,R] = (f16) src[R,C]
// ---------------------------------------------------------------------------
__global__ __launch_bounds__(256)
void transpose_f32_f16(const float* __restrict__ src, f16* __restrict__ dst, int R, int C)
{
    __shared__ float tile[32][33];
    const int cx = blockIdx.x * 32;
    const int ry = blockIdx.y * 32;
#pragma unroll
    for (int i = 0; i < 4; ++i) {
        const int r = ry + threadIdx.y + i * 8;
        const int c = cx + threadIdx.x;
        if (r < R && c < C) tile[threadIdx.y + i * 8][threadIdx.x] = src[(long long)r * C + c];
    }
    __syncthreads();
#pragma unroll
    for (int i = 0; i < 4; ++i) {
        const int c = cx + threadIdx.y + i * 8;
        const int r = ry + threadIdx.x;
        if (r < R && c < C) dst[(long long)c * R + r] = (f16)tile[threadIdx.x][threadIdx.y + i * 8];
    }
}

// ---------------------------------------------------------------------------
// Cross-entropy: one block per row; -log_softmax(logits)[target], mean-reduced
// via atomicAdd into *loss (must be zeroed beforehand).
// ---------------------------------------------------------------------------
__global__ __launch_bounds__(256)
void ce_loss(const float* __restrict__ logits, const int* __restrict__ targets,
             float* __restrict__ loss, float invN)
{
    const long long row = blockIdx.x;
    const float* p = logits + row * (long long)V_;
    __shared__ float red[256];
    const int tid = threadIdx.x;

    float mx = -1e30f;
    for (int c = tid; c < V_; c += 256) mx = fmaxf(mx, p[c]);
    red[tid] = mx; __syncthreads();
    for (int st = 128; st; st >>= 1) { if (tid < st) red[tid] = fmaxf(red[tid], red[tid + st]); __syncthreads(); }
    mx = red[0]; __syncthreads();

    float sum = 0.0f;
    for (int c = tid; c < V_; c += 256) sum += expf(p[c] - mx);
    red[tid] = sum; __syncthreads();
    for (int st = 128; st; st >>= 1) { if (tid < st) red[tid] += red[tid + st]; __syncthreads(); }

    if (tid == 0) {
        const int tgt = targets[row];
        const float lp = p[tgt] - mx - logf(red[0]);
        atomicAdd(loss, -lp * invN);
    }
}

// ---------------------------------------------------------------------------
extern "C" void kernel_launch(void* const* d_in, const int* in_sizes, int n_in,
                              void* d_out, int out_size, void* d_ws, size_t ws_size,
                              hipStream_t stream)
{
    (void)in_sizes; (void)n_in; (void)out_size; (void)ws_size;

    const int*   idx     = (const int*)  d_in[0];
    const int*   targets = (const int*)  d_in[1];
    const float* tok_emb = (const float*)d_in[2];
    const float* pos_emb = (const float*)d_in[3];
    const float* ln1_w   = (const float*)d_in[4];
    const float* ln1_b   = (const float*)d_in[5];
    const float* Wq      = (const float*)d_in[6];
    const float* Wk      = (const float*)d_in[7];
    const float* Wv      = (const float*)d_in[8];
    const float* proj_w  = (const float*)d_in[9];
    const float* proj_b  = (const float*)d_in[10];
    const float* ln2_w   = (const float*)d_in[11];
    const float* ln2_b   = (const float*)d_in[12];
    const float* ffn_w1  = (const float*)d_in[13];
    const float* ffn_b1  = (const float*)d_in[14];
    const float* ffn_w2  = (const float*)d_in[15];
    const float* ffn_b2  = (const float*)d_in[16];
    const float* lnf_w   = (const float*)d_in[17];
    const float* lnf_b   = (const float*)d_in[18];
    const float* lm_w    = (const float*)d_in[19];
    const float* lm_b    = (const float*)d_in[20];

    float* logits = (float*)d_out;
    float* lossp  = logits + (size_t)B_ * T_ * V_;

    // ---- workspace carve-out (all offsets 256B aligned) ----
    char* p = (char*)d_ws;
    auto carve = [&](size_t bytes) -> char* {
        char* r = p; p += (bytes + 255) & ~(size_t)255; return r;
    };
    float* x     = (float*)carve((size_t)B_ * T_ * E_ * 4);       // residual stream
    f16*   h16   = (f16*)  carve((size_t)B_ * T_ * E_ * 2);       // LN outputs
    f16*   tmp16 = (f16*)  carve((size_t)B_ * T_ * E_ * 2);       // GEMM f16 temp
    f16*   qh    = (f16*)  carve((size_t)B_ * H_ * T_ * D_ * 2);  // [B,H,T,D]
    f16*   kh    = (f16*)  carve((size_t)B_ * H_ * T_ * D_ * 2);  // [B,H,T,D]
    f16*   vt    = (f16*)  carve((size_t)B_ * H_ * D_ * T_ * 2);  // [B,H,D,T]
    f16*   att   = (f16*)  carve((size_t)B_ * H_ * T_ * T_ * 2);  // scores/att
    f16*   wtQ   = (f16*)  carve((size_t)E_ * E_ * 2);
    f16*   wtK   = (f16*)  carve((size_t)E_ * E_ * 2);
    f16*   wtV   = (f16*)  carve((size_t)E_ * E_ * 2);
    f16*   wtP   = (f16*)  carve((size_t)E_ * E_ * 2);
    f16*   wtF1  = (f16*)  carve((size_t)E_ * E_ * 2);
    f16*   wtF2  = (f16*)  carve((size_t)E_ * E_ * 2);
    f16*   lmT   = (f16*)  carve((size_t)V_ * E_ * 2);            // lm_w^T f16

    const long long BT = (long long)B_ * T_;
    const dim3 tb(32, 8);
    const dim3 tgE((E_ + 31) / 32, (E_ + 31) / 32);
    const unsigned elemGrid = (unsigned)((BT * E_ + 255) / 256);

    auto gemm = [&](const f16* A, const f16* Bt2, const float* bias, const float* res,
                    float* Cf, f16* Ch, int M, int N, int K, int ldc,
                    long long aB, long long bB, int czD, long long czO, long long czI,
                    int flags, int batch) {
        dim3 g((M + 127) / 128, (N + 127) / 128, batch);
        gemm_wmma_f16<<<g, 128, 0, stream>>>(A, Bt2, bias, res, Cf, Ch,
                                             M, N, K, ldc, aB, bB, czD, czO, czI, flags);
    };

    const float scale = 0.03125f;  // E^-0.5 = 1/32 (reference scales by full E)

    embed_mul<<<elemGrid, 256, 0, stream>>>(idx, tok_emb, pos_emb, x);

    for (int l = 0; l < L_; ++l) {
        const size_t wOff = (size_t)l * E_ * E_;
        const size_t bOff = (size_t)l * E_;

        transpose_f32_f16<<<tgE, tb, 0, stream>>>(Wq     + wOff, wtQ,  E_, E_);
        transpose_f32_f16<<<tgE, tb, 0, stream>>>(Wk     + wOff, wtK,  E_, E_);
        transpose_f32_f16<<<tgE, tb, 0, stream>>>(Wv     + wOff, wtV,  E_, E_);
        transpose_f32_f16<<<tgE, tb, 0, stream>>>(proj_w + wOff, wtP,  E_, E_);
        transpose_f32_f16<<<tgE, tb, 0, stream>>>(ffn_w1 + wOff, wtF1, E_, E_);
        transpose_f32_f16<<<tgE, tb, 0, stream>>>(ffn_w2 + wOff, wtF2, E_, E_);

        layernorm_f16<<<(unsigned)BT, 256, 0, stream>>>(x, ln1_w + bOff, ln1_b + bOff, h16);

        // q, k, v projections
        gemm(h16, wtQ, nullptr, nullptr, nullptr, tmp16, (int)BT, E_, E_, E_, 0, 0, 1, 0, 0, 0, 1);
        split_heads<<<elemGrid, 256, 0, stream>>>(tmp16, qh, 0);
        gemm(h16, wtK, nullptr, nullptr, nullptr, tmp16, (int)BT, E_, E_, E_, 0, 0, 1, 0, 0, 0, 1);
        split_heads<<<elemGrid, 256, 0, stream>>>(tmp16, kh, 0);
        gemm(h16, wtV, nullptr, nullptr, nullptr, tmp16, (int)BT, E_, E_, E_, 0, 0, 1, 0, 0, 0, 1);
        split_heads<<<elemGrid, 256, 0, stream>>>(tmp16, vt, 1);

        // scores = q @ k^T  (batched over B*H), stored f16 in att
        gemm(qh, kh, nullptr, nullptr, nullptr, att, T_, T_, D_, T_,
             (long long)T_ * D_, (long long)T_ * D_, 1, (long long)T_ * T_, 0, 0, B_ * H_);
        softmax_causal<<<(unsigned)(B_ * H_ * T_), 256, 0, stream>>>(att, scale);

        // o = att @ v, written directly in merged [B*T, E] layout
        gemm(att, vt, nullptr, nullptr, nullptr, tmp16, T_, D_, T_, E_,
             (long long)T_ * T_, (long long)D_ * T_, H_, (long long)T_ * E_, D_, 0, B_ * H_);

        // x = x + o @ proj_w + proj_b
        gemm(tmp16, wtP, proj_b + bOff, x, x, nullptr, (int)BT, E_, E_, E_,
             0, 0, 1, 0, 0, GF_BIAS | GF_RES, 1);

        layernorm_f16<<<(unsigned)BT, 256, 0, stream>>>(x, ln2_w + bOff, ln2_b + bOff, h16);

        // ffn: x = x + relu(h2 @ W1 + b1) @ W2 + b2
        gemm(h16, wtF1, ffn_b1 + bOff, nullptr, nullptr, tmp16, (int)BT, E_, E_, E_,
             0, 0, 1, 0, 0, GF_BIAS | GF_RELU, 1);
        gemm(tmp16, wtF2, ffn_b2 + bOff, x, x, nullptr, (int)BT, E_, E_, E_,
             0, 0, 1, 0, 0, GF_BIAS | GF_RES, 1);
    }

    layernorm_f16<<<(unsigned)BT, 256, 0, stream>>>(x, lnf_w, lnf_b, h16);

    // LM head: logits = h @ lm_w + lm_b  (N = 50257, ragged edge handled in GEMM)
    transpose_f32_f16<<<dim3((V_ + 31) / 32, (E_ + 31) / 32), tb, 0, stream>>>(lm_w, lmT, E_, V_);
    gemm(h16, lmT, lm_b, nullptr, logits, nullptr, (int)BT, V_, E_, V_,
         0, 0, 1, 0, 0, GF_BIAS, 1);

    hipMemsetAsync(lossp, 0, sizeof(float), stream);
    ce_loss<<<(unsigned)BT, 256, 0, stream>>>(logits, targets, lossp, 1.0f / (float)BT);
}
